// TransformerEncoder_91010357002585
// MI455X (gfx1250) — matrix operand order repaired and checked by reading, not compiled, not run on hardware
//
#include <hip/hip_runtime.h>
#include <hip/hip_bf16.h>
#include <math.h>

#define BATCH   8
#define SEQ     1024
#define DIM     512
#define HEADS   8
#define DHEAD   64
#define FFDIM   2048
#define LAYERS  6
#define TOKENS  (BATCH * SEQ)

typedef __attribute__((ext_vector_type(16))) __bf16 v16bf;
typedef __attribute__((ext_vector_type(8)))  float  v8f;

struct alignas(16) U4 { unsigned int w[4]; };
union FragBF { U4 q[2]; v16bf f; };

__device__ __forceinline__ unsigned short f2bf(float x) {
  unsigned int u = __float_as_uint(x);
  unsigned int r = u + 0x7FFFu + ((u >> 16) & 1u);   // round-to-nearest-even
  return (unsigned short)(r >> 16);
}

// ---------------------------------------------------------------------------
// positions[b][s] = cumsum(mask != 0) * (mask != 0)   (block-wide scan, B blocks)
// ---------------------------------------------------------------------------
__global__ __launch_bounds__(1024) void pos_scan_kernel(const int* __restrict__ mask,
                                                        int* __restrict__ pos) {
  __shared__ int sm[SEQ];
  const int b = blockIdx.x, t = threadIdx.x;
  const int v = (mask[b * SEQ + t] != 0) ? 1 : 0;
  sm[t] = v;
  __syncthreads();
  for (int d = 1; d < SEQ; d <<= 1) {
    int x = (t >= d) ? sm[t - d] : 0;
    __syncthreads();
    sm[t] += x;
    __syncthreads();
  }
  pos[b * SEQ + t] = sm[t] * v;
}

// ---------------------------------------------------------------------------
// x += sinusoidal_table[pos]; write fp32 + bf16 copies. grid = TOKENS, block 256.
// ---------------------------------------------------------------------------
__global__ __launch_bounds__(256) void pos_add_kernel(const float* __restrict__ x,
                                                      const int* __restrict__ pos,
                                                      float* __restrict__ xf,
                                                      unsigned short* __restrict__ xh) {
  const int token = blockIdx.x;
  const int p = pos[token];
  const size_t base = (size_t)token * DIM;
#pragma unroll
  for (int j = 0; j < 2; ++j) {
    const int dd = threadIdx.x + j * 256;
    float add = 0.0f;
    if (p != 0) {
      const int i = (dd < 256) ? dd : dd - 256;
      const float freq = __expf(-logf(10000.0f) * (float)i / 255.0f);
      const float ang  = (float)p * freq;
      add = (dd < 256) ? sinf(ang) : cosf(ang);
    }
    const float v = x[base + dd] + add;
    xf[base + dd] = v;
    xh[base + dd] = f2bf(v);
  }
}

// ---------------------------------------------------------------------------
// W [K,N] fp32  ->  Wt [N,K] bf16 (per layer via blockIdx.z). block (32,8).
// ---------------------------------------------------------------------------
__global__ __launch_bounds__(256) void wconv_kernel(const float* __restrict__ W,
                                                    unsigned short* __restrict__ Wt,
                                                    int K, int N) {
  __shared__ float tile[32][33];
  const float* Wl = W + (size_t)blockIdx.z * K * N;
  unsigned short* Wtl = Wt + (size_t)blockIdx.z * K * N;
  const int n0 = blockIdx.x * 32, k0 = blockIdx.y * 32;
  const int tx = threadIdx.x, ty = threadIdx.y;
#pragma unroll
  for (int i = 0; i < 32; i += 8)
    tile[ty + i][tx] = Wl[(size_t)(k0 + ty + i) * N + (n0 + tx)];
  __syncthreads();
#pragma unroll
  for (int i = 0; i < 32; i += 8)
    Wtl[(size_t)(n0 + ty + i) * K + (k0 + tx)] = f2bf(tile[tx][ty + i]);
}

// ---------------------------------------------------------------------------
// Generic bf16 WMMA GEMM:  C[M,N] = A[M,K] * Bt[N,K]^T  (+bias, relu)
// Block 256 = 8 waves; wave tile 32x64 (2x4 accumulators), block tile 128x128.
// qkvMode: scatter output to q[b,h,s,d], k[b,h,s,d], vT[b,h,d,s] (bf16).
// ---------------------------------------------------------------------------
__global__ __launch_bounds__(256) void gemm_bf16_wmma(
    const unsigned short* __restrict__ A,
    const unsigned short* __restrict__ Bt,
    const float* __restrict__ bias,
    float* __restrict__ outF,
    unsigned short* __restrict__ outH,
    int M, int N, int K, int doRelu, int qkvMode,
    unsigned short* __restrict__ qOut,
    unsigned short* __restrict__ kOut,
    unsigned short* __restrict__ vtOut) {
  const int tid = threadIdx.x;
  const int w   = tid >> 5;
  const int l   = tid & 31;
  const int hlf = l >> 4;     // 0: lanes 0-15, 1: lanes 16-31
  const int ln  = l & 15;
  const int rowBase = blockIdx.y * 128 + (w & 3) * 32;
  const int colBase = blockIdx.x * 128 + (w >> 2) * 64;

  v8f acc[2][4];
#pragma unroll
  for (int mt = 0; mt < 2; ++mt)
#pragma unroll
    for (int nt = 0; nt < 4; ++nt)
#pragma unroll
      for (int r = 0; r < 8; ++r) acc[mt][nt][r] = 0.0f;

  for (int k0 = 0; k0 < K; k0 += 32) {
    FragBF a[2], b[4];
#pragma unroll
    for (int mt = 0; mt < 2; ++mt) {
      // A frag: lane<16 holds K {0..7,16..23}; lane>=16 holds K {8..15,24..31}
      const unsigned short* p = A + (size_t)(rowBase + mt * 16 + ln) * K + k0 + hlf * 8;
      a[mt].q[0] = *(const U4*)p;
      a[mt].q[1] = *(const U4*)(p + 16);
    }
#pragma unroll
    for (int nt = 0; nt < 4; ++nt) {
      // B frag: lane<16 holds K {0..7,16..23}; lane>=16 holds K {8..15,24..31}
      const unsigned short* p = Bt + (size_t)(colBase + nt * 16 + ln) * K + k0 + hlf * 8;
      b[nt].q[0] = *(const U4*)p;
      b[nt].q[1] = *(const U4*)(p + 16);
    }
    if (k0 + 32 < K) {
      __builtin_prefetch(A + (size_t)(rowBase + ln) * K + k0 + 32, 0, 1);
      __builtin_prefetch(Bt + (size_t)(colBase + ln) * K + k0 + 32, 0, 1);
    }
#pragma unroll
    for (int mt = 0; mt < 2; ++mt)
#pragma unroll
      for (int nt = 0; nt < 4; ++nt)
        acc[mt][nt] = __builtin_amdgcn_wmma_f32_16x16x32_bf16(
            false, a[mt].f, false, b[nt].f, (short)0, acc[mt][nt], false, false);
  }

#pragma unroll
  for (int mt = 0; mt < 2; ++mt) {
#pragma unroll
    for (int nt = 0; nt < 4; ++nt) {
      const int col = colBase + nt * 16 + ln;
#pragma unroll
      for (int r = 0; r < 8; ++r) {
        const int row = rowBase + mt * 16 + hlf * 8 + r;
        float v = acc[mt][nt][r];
        if (!qkvMode) {
          if (bias)   v += bias[col];
          if (doRelu) v  = fmaxf(v, 0.0f);
          if (outF) outF[(size_t)row * N + col] = v;
          if (outH) outH[(size_t)row * N + col] = f2bf(v);
        } else {
          const int bb = row >> 10, ss = row & 1023;
          const unsigned short hv = f2bf(v);
          if (col < DIM) {
            const int hh = col >> 6, dd = col & 63;
            qOut[(((size_t)bb * HEADS + hh) * SEQ + ss) * DHEAD + dd] = hv;
          } else if (col < 2 * DIM) {
            const int c2 = col - DIM, hh = c2 >> 6, dd = c2 & 63;
            kOut[(((size_t)bb * HEADS + hh) * SEQ + ss) * DHEAD + dd] = hv;
          } else {
            const int c2 = col - 2 * DIM, hh = c2 >> 6, dd = c2 & 63;
            vtOut[(((size_t)bb * HEADS + hh) * DHEAD + dd) * SEQ + ss] = hv;
          }
        }
      }
    }
  }
}

// ---------------------------------------------------------------------------
// Flash attention, one (b,h,128-query-block) per workgroup; 8 waves x 16 rows.
// scale = 1 (reference uses unscaled QK^T).
// ---------------------------------------------------------------------------
__global__ __launch_bounds__(256) void attn_kernel(
    const unsigned short* __restrict__ qb,
    const unsigned short* __restrict__ kb,
    const unsigned short* __restrict__ vt,
    const int* __restrict__ mask,
    unsigned short* __restrict__ attnOut) {
  __shared__ alignas(16) unsigned short pls[8][16 * 64];  // per-wave P tile (bf16)

  const int tid = threadIdx.x;
  const int w   = tid >> 5;
  const int l   = tid & 31;
  const int hlf = l >> 4;
  const int ln  = l & 15;

  const int blk  = blockIdx.x;       // B*H*(S/128) = 512
  const int bh   = blk >> 3;
  const int qblk = blk & 7;
  const int b    = bh >> 3;
  const int hh   = bh & 7;

  const unsigned short* Q  = qb + (size_t)bh * SEQ * DHEAD;
  const unsigned short* Kp = kb + (size_t)bh * SEQ * DHEAD;
  const unsigned short* Vp = vt + (size_t)bh * DHEAD * SEQ;
  const int qrow0 = qblk * 128 + w * 16;

  // Preload Q A-frags (dh = 64 -> two K-steps of 32)
  FragBF qf[2];
#pragma unroll
  for (int kk = 0; kk < 2; ++kk) {
    const unsigned short* p = Q + (size_t)(qrow0 + ln) * DHEAD + kk * 32 + hlf * 8;
    qf[kk].q[0] = *(const U4*)p;
    qf[kk].q[1] = *(const U4*)(p + 16);
  }

  v8f o[4];
  float mOld[8], lSum[8];
#pragma unroll
  for (int dt = 0; dt < 4; ++dt)
#pragma unroll
    for (int r = 0; r < 8; ++r) o[dt][r] = 0.0f;
#pragma unroll
  for (int r = 0; r < 8; ++r) { mOld[r] = -1.0e30f; lSum[r] = 0.0f; }

  for (int kbk = 0; kbk < SEQ / 64; ++kbk) {
    // ---- scores S = Q K^T for 16 rows x 64 keys ----
    v8f s[4];
#pragma unroll
    for (int c = 0; c < 4; ++c)
#pragma unroll
      for (int r = 0; r < 8; ++r) s[c][r] = 0.0f;

#pragma unroll
    for (int c = 0; c < 4; ++c) {
      FragBF bk[2];
#pragma unroll
      for (int kk = 0; kk < 2; ++kk) {
        const unsigned short* p =
            Kp + (size_t)(kbk * 64 + c * 16 + ln) * DHEAD + kk * 32 + hlf * 8;
        bk[kk].q[0] = *(const U4*)p;
        bk[kk].q[1] = *(const U4*)(p + 16);
      }
      s[c] = __builtin_amdgcn_wmma_f32_16x16x32_bf16(false, qf[0].f, false, bk[0].f,
                                                     (short)0, s[c], false, false);
      s[c] = __builtin_amdgcn_wmma_f32_16x16x32_bf16(false, qf[1].f, false, bk[1].f,
                                                     (short)0, s[c], false, false);
    }

    // ---- key mask: column = key index (lane-resident) ----
#pragma unroll
    for (int c = 0; c < 4; ++c) {
      const int key = kbk * 64 + c * 16 + ln;
      if (mask[b * SEQ + key] == 0) {
#pragma unroll
        for (int r = 0; r < 8; ++r) s[c][r] = -1.0e30f;
      }
    }

    // ---- online softmax row stats (row lives across 16 lanes of a half) ----
    float mNew[8], alpha[8];
#pragma unroll
    for (int r = 0; r < 8; ++r) {
      float mx = s[0][r];
#pragma unroll
      for (int c = 1; c < 4; ++c) mx = fmaxf(mx, s[c][r]);
#pragma unroll
      for (int d = 1; d < 16; d <<= 1) mx = fmaxf(mx, __shfl_xor(mx, d, 32));
      mNew[r]  = fmaxf(mOld[r], mx);
      alpha[r] = __expf(mOld[r] - mNew[r]);
    }
#pragma unroll
    for (int r = 0; r < 8; ++r) {
      float sum = 0.0f;
#pragma unroll
      for (int c = 0; c < 4; ++c) {
        const float pv = __expf(s[c][r] - mNew[r]);
        s[c][r] = pv;
        sum += pv;
      }
#pragma unroll
      for (int d = 1; d < 16; d <<= 1) sum += __shfl_xor(sum, d, 32);
      lSum[r] = lSum[r] * alpha[r] + sum;
      mOld[r] = mNew[r];
    }
#pragma unroll
    for (int dt = 0; dt < 4; ++dt)
#pragma unroll
      for (int r = 0; r < 8; ++r) o[dt][r] *= alpha[r];

    // ---- P (C-layout) -> bf16 LDS -> A-layout frags ----
    unsigned short* pw = pls[w];
#pragma unroll
    for (int c = 0; c < 4; ++c)
#pragma unroll
      for (int r = 0; r < 8; ++r)
        pw[(hlf * 8 + r) * 64 + c * 16 + ln] = f2bf(s[c][r]);
    __syncthreads();

    FragBF pf[2];
#pragma unroll
    for (int kk = 0; kk < 2; ++kk) {
      const unsigned short* pp = pw + (size_t)ln * 64 + kk * 32 + hlf * 8;
      pf[kk].q[0] = *(const U4*)pp;
      pf[kk].q[1] = *(const U4*)(pp + 16);
    }

    // ---- O += P V  (B-frag from V^T rows = output dims) ----
#pragma unroll
    for (int dt = 0; dt < 4; ++dt) {
      FragBF vf[2];
#pragma unroll
      for (int kk = 0; kk < 2; ++kk) {
        const unsigned short* p =
            Vp + (size_t)(dt * 16 + ln) * SEQ + kbk * 64 + kk * 32 + hlf * 8;
        vf[kk].q[0] = *(const U4*)p;
        vf[kk].q[1] = *(const U4*)(p + 16);
      }
      o[dt] = __builtin_amdgcn_wmma_f32_16x16x32_bf16(false, pf[0].f, false, vf[0].f,
                                                      (short)0, o[dt], false, false);
      o[dt] = __builtin_amdgcn_wmma_f32_16x16x32_bf16(false, pf[1].f, false, vf[1].f,
                                                      (short)0, o[dt], false, false);
    }
    __syncthreads();
  }

  // ---- normalize + store to [token][h*64+d] bf16 ----
#pragma unroll
  for (int dt = 0; dt < 4; ++dt) {
    const int col = hh * DHEAD + dt * 16 + ln;
#pragma unroll
    for (int r = 0; r < 8; ++r) {
      const int srow  = qrow0 + hlf * 8 + r;
      const int token = b * SEQ + srow;
      attnOut[(size_t)token * DIM + col] = f2bf(o[dt][r] / lSum[r]);
    }
  }
}

// ---------------------------------------------------------------------------
// y = LN(gin + resid) * g + b ; writes fp32 + bf16. One token per block (256 thr).
// ---------------------------------------------------------------------------
__global__ __launch_bounds__(256) void ln_kernel(const float* __restrict__ gin,
                                                 const float* __restrict__ resid,
                                                 const float* __restrict__ gg,
                                                 const float* __restrict__ bb,
                                                 float* __restrict__ xfOut,
                                                 unsigned short* __restrict__ xhOut) {
  __shared__ float red[256];
  const int token = blockIdx.x, t = threadIdx.x;
  const size_t base = (size_t)token * DIM;
  const float v0 = gin[base + t] + resid[base + t];
  const float v1 = gin[base + t + 256] + resid[base + t + 256];

  red[t] = v0 + v1;
  __syncthreads();
  for (int o2 = 128; o2 > 0; o2 >>= 1) {
    if (t < o2) red[t] += red[t + o2];
    __syncthreads();
  }
  const float mu = red[0] * (1.0f / (float)DIM);
  __syncthreads();

  const float d0 = v0 - mu, d1 = v1 - mu;
  red[t] = d0 * d0 + d1 * d1;
  __syncthreads();
  for (int o2 = 128; o2 > 0; o2 >>= 1) {
    if (t < o2) red[t] += red[t + o2];
    __syncthreads();
  }
  const float rstd = rsqrtf(red[0] * (1.0f / (float)DIM) + 1e-5f);

  const float y0 = d0 * rstd * gg[t] + bb[t];
  const float y1 = d1 * rstd * gg[t + 256] + bb[t + 256];
  xfOut[base + t] = y0;
  xhOut[base + t] = f2bf(y0);
  xfOut[base + t + 256] = y1;
  xhOut[base + t + 256] = f2bf(y1);
}

// ---------------------------------------------------------------------------
// Host-side orchestration
// ---------------------------------------------------------------------------
extern "C" void kernel_launch(void* const* d_in, const int* in_sizes, int n_in,
                              void* d_out, int out_size, void* d_ws, size_t ws_size,
                              hipStream_t stream) {
  (void)in_sizes; (void)n_in; (void)ws_size;
  const float* x     = (const float*)d_in[0];
  // d_in[1] char_ids, d_in[2] seq_len: unused by the reference
  const int*   mask  = (const int*)d_in[3];
  const float* Wqkv  = (const float*)d_in[4];
  const float* Wfc   = (const float*)d_in[5];
  const float* bfc   = (const float*)d_in[6];
  const float* ln1g  = (const float*)d_in[7];
  const float* ln1b  = (const float*)d_in[8];
  const float* ln2g  = (const float*)d_in[9];
  const float* ln2b  = (const float*)d_in[10];
  const float* W1    = (const float*)d_in[11];
  const float* b1    = (const float*)d_in[12];
  const float* W2    = (const float*)d_in[13];
  const float* b2    = (const float*)d_in[14];

  char* ws = (char*)d_ws;
  size_t off = 0;
  auto take = [&](size_t bytes) -> void* {
    void* p = ws + off;
    off = (off + bytes + 255) & ~(size_t)255;
    return p;
  };

  int*            posn  = (int*)take((size_t)TOKENS * 4);
  float*          xf    = (float*)take((size_t)TOKENS * DIM * 4);
  unsigned short* xh    = (unsigned short*)take((size_t)TOKENS * DIM * 2);
  unsigned short* qbuf  = (unsigned short*)take((size_t)TOKENS * DIM * 2);
  unsigned short* kbuf  = (unsigned short*)take((size_t)TOKENS * DIM * 2);
  unsigned short* vtbuf = (unsigned short*)take((size_t)TOKENS * DIM * 2);
  unsigned short* attnO = (unsigned short*)take((size_t)TOKENS * DIM * 2);
  unsigned short* hbuf  = (unsigned short*)take((size_t)TOKENS * FFDIM * 2);
  float*          gout  = (float*)take((size_t)TOKENS * DIM * 4);
  unsigned short* wqkvt = (unsigned short*)take((size_t)LAYERS * DIM * 3 * DIM * 2);
  unsigned short* wfct  = (unsigned short*)take((size_t)LAYERS * DIM * DIM * 2);
  unsigned short* w1t   = (unsigned short*)take((size_t)LAYERS * DIM * FFDIM * 2);
  unsigned short* w2t   = (unsigned short*)take((size_t)LAYERS * FFDIM * DIM * 2);

  // positions + positional embedding
  pos_scan_kernel<<<BATCH, SEQ, 0, stream>>>(mask, posn);
  pos_add_kernel<<<TOKENS, 256, 0, stream>>>(x, posn, xf, xh);

  // weight transpose + bf16 convert (all layers via grid.z)
  wconv_kernel<<<dim3(3 * DIM / 32, DIM / 32, LAYERS), dim3(32, 8), 0, stream>>>(
      Wqkv, wqkvt, DIM, 3 * DIM);
  wconv_kernel<<<dim3(DIM / 32, DIM / 32, LAYERS), dim3(32, 8), 0, stream>>>(
      Wfc, wfct, DIM, DIM);
  wconv_kernel<<<dim3(FFDIM / 32, DIM / 32, LAYERS), dim3(32, 8), 0, stream>>>(
      W1, w1t, DIM, FFDIM);
  wconv_kernel<<<dim3(DIM / 32, FFDIM / 32, LAYERS), dim3(32, 8), 0, stream>>>(
      W2, w2t, FFDIM, DIM);

  const dim3 blk256(256);
  for (int l = 0; l < LAYERS; ++l) {
    // QKV projection (scatter to q/k/vT)
    gemm_bf16_wmma<<<dim3(3 * DIM / 128, TOKENS / 128), blk256, 0, stream>>>(
        xh, wqkvt + (size_t)l * DIM * 3 * DIM, nullptr, nullptr, nullptr,
        TOKENS, 3 * DIM, DIM, 0, 1, qbuf, kbuf, vtbuf);

    // attention
    attn_kernel<<<BATCH * HEADS * (SEQ / 128), blk256, 0, stream>>>(
        qbuf, kbuf, vtbuf, mask, attnO);

    // output projection + bias
    gemm_bf16_wmma<<<dim3(DIM / 128, TOKENS / 128), blk256, 0, stream>>>(
        attnO, wfct + (size_t)l * DIM * DIM, bfc + (size_t)l * DIM, gout, nullptr,
        TOKENS, DIM, DIM, 0, 0, nullptr, nullptr, nullptr);

    // LN1 (residual = xf), overwrite xf/xh
    ln_kernel<<<TOKENS, blk256, 0, stream>>>(gout, xf, ln1g + (size_t)l * DIM,
                                             ln1b + (size_t)l * DIM, xf, xh);

    // FFN up + ReLU -> bf16 h
    gemm_bf16_wmma<<<dim3(FFDIM / 128, TOKENS / 128), blk256, 0, stream>>>(
        xh, w1t + (size_t)l * DIM * FFDIM, b1 + (size_t)l * FFDIM, nullptr, hbuf,
        TOKENS, FFDIM, DIM, 1, 0, nullptr, nullptr, nullptr);

    // FFN down + bias
    gemm_bf16_wmma<<<dim3(DIM / 128, TOKENS / 128), blk256, 0, stream>>>(
        hbuf, w2t + (size_t)l * FFDIM * DIM, b2 + (size_t)l * DIM, gout, nullptr,
        TOKENS, DIM, FFDIM, 0, 0, nullptr, nullptr, nullptr);

    // LN2 (residual = xf)
    ln_kernel<<<TOKENS, blk256, 0, stream>>>(gout, xf, ln2g + (size_t)l * DIM,
                                             ln2b + (size_t)l * DIM, xf, xh);
  }

  hipMemcpyAsync(d_out, xf, (size_t)out_size * sizeof(float),
                 hipMemcpyDeviceToDevice, stream);
}
